// LatentResonanceAttentionV18_73134703116764
// MI455X (gfx1250) — matrix operand order repaired and checked by reading, not compiled, not run on hardware
//
#include <hip/hip_runtime.h>
#include <hip/hip_bf16.h>
#include <math.h>

typedef __attribute__((ext_vector_type(16))) __bf16 v16bf;
typedef __attribute__((ext_vector_type(8)))  float  v8f;

#define D_    768
#define H_    12
#define DH_   64
#define LAT_  512
#define B_    2
#define L_    1024
#define MROWS (B_*L_)

#if defined(__has_builtin)
#if __has_builtin(__builtin_amdgcn_global_load_async_to_lds_b128)
#define USE_ASYNC_LDS 1
#endif
#endif
#ifndef USE_ASYNC_LDS
#define USE_ASYNC_LDS 0
#endif

#if USE_ASYNC_LDS
// expected param types per hipcc diagnostic:
//   arg0: __attribute__((vector_size(16))) int __device__ *   (AS1 / global)
//   arg1: (by symmetry) same pointee in __shared__ (AS3)
typedef int v4i_async __attribute__((vector_size(4 * sizeof(int))));
#define ASYNC_GPTR(p) ((__attribute__((address_space(1))) v4i_async*)(p))
#define ASYNC_LPTR(p) ((__attribute__((address_space(3))) v4i_async*)(p))
#endif

__device__ __forceinline__ float sigmoidf_(float x) { return 1.0f / (1.0f + __expf(-x)); }
__device__ __forceinline__ float siluf_(float x)    { return x * sigmoidf_(x); }

// ---------------- weight cast f32 -> bf16 ----------------
__global__ void cast_f32_bf16(const float* __restrict__ src, __bf16* __restrict__ dst, int n) {
    int i = blockIdx.x * blockDim.x + threadIdx.x;
    if (i < n) dst[i] = (__bf16)src[i];
}

// ---------------- row RMSNorm -> bf16 ----------------
__global__ __launch_bounds__(256)
void rms_cast(const float* __restrict__ x, const float* __restrict__ w,
              __bf16* __restrict__ out, int ncol) {
    const int row = blockIdx.x;
    const int tid = threadIdx.x;
    __shared__ float red[256];
    const float* xp = x + (size_t)row * ncol;
    float ss = 0.f;
    for (int c = tid; c < ncol; c += 256) { float v = xp[c]; ss += v * v; }
    red[tid] = ss;
    __syncthreads();
    for (int s = 128; s > 0; s >>= 1) {
        if (tid < s) red[tid] += red[tid + s];
        __syncthreads();
    }
    const float r = rsqrtf(red[0] / (float)ncol + 1e-6f);
    __bf16* op = out + (size_t)row * ncol;
    for (int c = tid; c < ncol; c += 256) op[c] = (__bf16)(w[c] * xp[c] * r);
}

// ---------------- generic WMMA bf16 GEMM, 128x64 block tile, K-step 32 ----------------
// mode 0: outF = acc ; 1: outH = bf16(silu) ; 2: outF = silu ; 3: outF = acc*aux
__global__ __launch_bounds__(256)
void gemm_wmma_bf16(const __bf16* __restrict__ A, const __bf16* __restrict__ Bw,
                    float* __restrict__ outF, __bf16* __restrict__ outH,
                    const float* __restrict__ aux,
                    int M, int N, int K, int mode) {
    __shared__ __align__(16) __bf16 As[128 * 48];   // 128 rows x 32 K (stride 48: 96B, 16B-aligned chunks)
    __shared__ __align__(16) __bf16 BsT[64 * 48];   // transposed: 64 cols x 32 K (stride 48)
    const int tid   = threadIdx.x;
    const int wave  = tid >> 5;
    const int lane  = tid & 31;
    const int lhalf = lane >> 4;     // 0/1
    const int llow  = lane & 15;
    const int m0 = blockIdx.x * 128;
    const int n0 = blockIdx.y * 64;
    const unsigned short* Bu = (const unsigned short*)Bw;

    // staging maps
    const int ar  = tid >> 1;          // A: row 0..127
    const int ac0 = (tid & 1) * 16;    // A: 16-half chunk
    const int bcol = tid & 63;         // B: column within tile
    const int bk   = (tid >> 6) * 8;   // B: 8 consecutive K values

    v8f acc[4];
#pragma unroll
    for (int c = 0; c < 4; ++c) acc[c] = v8f{};

    for (int k0 = 0; k0 < K; k0 += 32) {
        __syncthreads();

        // ---- stage A tile (row-major) : 32 bytes per thread ----
        {
            const __bf16* src = A + (size_t)(m0 + ar) * K + k0 + ac0;
            __bf16* dst = &As[ar * 48 + ac0];
#if USE_ASYNC_LDS
            __builtin_amdgcn_global_load_async_to_lds_b128(
                ASYNC_GPTR(src), ASYNC_LPTR(dst), 0, 0);
            __builtin_amdgcn_global_load_async_to_lds_b128(
                ASYNC_GPTR(src + 8), ASYNC_LPTR(dst + 8), 0, 0);
#else
#pragma unroll
            for (int j = 0; j < 16; ++j) dst[j] = src[j];
#endif
        }

        // ---- stage B tile transposed: gather 8 K-values of one column, one b128 store ----
        {
            const size_t base = (size_t)(k0 + bk) * N + n0 + bcol;
            unsigned short t0 = Bu[base];
            unsigned short t1 = Bu[base + (size_t)N];
            unsigned short t2 = Bu[base + (size_t)2 * N];
            unsigned short t3 = Bu[base + (size_t)3 * N];
            unsigned short t4 = Bu[base + (size_t)4 * N];
            unsigned short t5 = Bu[base + (size_t)5 * N];
            unsigned short t6 = Bu[base + (size_t)6 * N];
            unsigned short t7 = Bu[base + (size_t)7 * N];
            uint4 payload;
            payload.x = (unsigned)t0 | ((unsigned)t1 << 16);
            payload.y = (unsigned)t2 | ((unsigned)t3 << 16);
            payload.z = (unsigned)t4 | ((unsigned)t5 << 16);
            payload.w = (unsigned)t6 | ((unsigned)t7 << 16);
            *(uint4*)&BsT[bcol * 48 + bk] = payload;
        }

        if (k0 + 32 < K) {   // CDNA5 global_prefetch_b8 of next K tile
            __builtin_prefetch(A + (size_t)(m0 + ar) * K + (k0 + 32), 0, 0);
            __builtin_prefetch(Bw + (size_t)(k0 + 32 + bk) * N + n0 + bcol, 0, 0);
        }

#if USE_ASYNC_LDS
#if __has_builtin(__builtin_amdgcn_s_wait_asynccnt)
        __builtin_amdgcn_s_wait_asynccnt(0);
#else
        asm volatile("s_wait_asynccnt 0x0" ::: "memory");
#endif
#endif
        __syncthreads();

        // A fragment: lane = row (wave*16+llow); K order matches B:
        // lanes 0-15: K = 0..15; lanes 16-31: K = 16..31
        // -> 16 contiguous halves: ds_load_b128 x2
        v16bf afrag;
        const __bf16* ap = &As[(wave * 16 + llow) * 48];
#pragma unroll
        for (int j = 0; j < 16; ++j)
            afrag[j] = ap[16 * lhalf + j];

#pragma unroll
        for (int c = 0; c < 4; ++c) {
            // B fragment: lane = col (c*16+llow); lanes 0-15: K=j, lanes 16-31: K=16+j
            // -> 16 contiguous halves in BsT: ds_load_b128 x2
            v16bf bfrag;
            const __bf16* bp = &BsT[(c * 16 + llow) * 48 + lhalf * 16];
#pragma unroll
            for (int j = 0; j < 16; ++j) bfrag[j] = bp[j];
            acc[c] = __builtin_amdgcn_wmma_f32_16x16x32_bf16(
                false, afrag, false, bfrag, (short)0, acc[c], false, false);
        }
    }

    // C/D layout: VGPR v, lanes 0-15 -> M=v, lanes 16-31 -> M=8+v
#pragma unroll
    for (int c = 0; c < 4; ++c) {
#pragma unroll
        for (int v = 0; v < 8; ++v) {
            const int row = m0 + wave * 16 + v + 8 * lhalf;
            const int col = n0 + c * 16 + llow;
            const size_t o = (size_t)row * N + col;
            const float val = acc[c][v];
            if (mode == 0)      outF[o] = val;
            else if (mode == 1) outH[o] = (__bf16)siluf_(val);
            else if (mode == 2) outF[o] = siluf_(val);
            else                outF[o] = val * aux[o];
        }
    }
}

// ---------------- reso params + gate (scalar, tiny N) ----------------
__global__ __launch_bounds__(256)
void gate_kernel(const __bf16* __restrict__ lat, const float* __restrict__ w_reso,
                 const float* __restrict__ temperature, float* __restrict__ gate) {
    const int g = blockIdx.x * blockDim.x + threadIdx.x;
    if (g >= MROWS * H_) return;
    const int h   = g % H_;
    const int row = g / H_;
    const __bf16* lp = lat + (size_t)row * LAT_;
    float p0 = 0.f, p1 = 0.f, p2 = 0.f, p3 = 0.f;
    for (int l = 0; l < LAT_; ++l) {
        const float lv = (float)lp[l];
        const float4 wr = *(const float4*)(w_reso + (size_t)l * (H_ * 4) + h * 4);
        p0 += lv * wr.x; p1 += lv * wr.y; p2 += lv * wr.z; p3 += lv * wr.w;
    }
    const float pi = 3.14159265358979323846f;
    const float sp = sigmoidf_(p1) * pi;
    const float cp = sigmoidf_(p3) * pi;
    const float val = sigmoidf_(sigmoidf_(p0) * sigmoidf_(p2) * __cosf(sp - cp) * temperature[0]) * 1.2f - 0.1f;
    gate[(size_t)row * H_ + h] = fminf(fmaxf(val, 0.05f), 0.95f);
}

// ---------------- per-(b,l,h) head RMS + RoPE + elu+1, in place ----------------
__global__ __launch_bounds__(64)
void qk_transform(float* __restrict__ qkv, const float* __restrict__ w_qn,
                  const float* __restrict__ w_kn) {
    const int idx = blockIdx.x;          // row * H + h
    const int h   = idx % H_;
    const int row = idx / H_;
    const int t   = row % L_;            // sequence position
    const int i   = threadIdx.x;         // dh index

    float* qp = qkv + (size_t)row * (3 * D_) + h * DH_;
    float* kp = qp + D_;

    __shared__ __align__(16) float sq[DH_];
    __shared__ __align__(16) float sk[DH_];
    const float qv = qp[i];
    const float kv = kp[i];
    sq[i] = qv; sk[i] = kv;
    __syncthreads();

    float ssq = 0.f, ssk = 0.f;
    const float4* sq4 = (const float4*)sq;
    const float4* sk4 = (const float4*)sk;
#pragma unroll
    for (int j = 0; j < DH_ / 4; ++j) {
        const float4 a = sq4[j], b = sk4[j];
        ssq += a.x * a.x + a.y * a.y + a.z * a.z + a.w * a.w;
        ssk += b.x * b.x + b.y * b.y + b.z * b.z + b.w * b.w;
    }
    const float rq = rsqrtf(ssq / (float)DH_ + 1e-6f);
    const float rk = rsqrtf(ssk / (float)DH_ + 1e-6f);

    const int   p   = (i < 32) ? i + 32 : i - 32;
    const float sgn = (i < 32) ? -1.f : 1.f;
    const float qn  = w_qn[i] * qv * rq;
    const float qnp = w_qn[p] * sq[p] * rq;
    const float kn  = w_kn[i] * kv * rk;
    const float knp = w_kn[p] * sk[p] * rk;

    const int j = i & 31;
    const float inv = __expf(-(float)j * (9.210340371976184f / 32.0f)); // 10000^(-2j/64)
    float s, c;
    __sincosf((float)t * inv, &s, &c);

    const float qr = qn * c + sgn * qnp * s;
    const float kr = kn * c + sgn * knp * s;

    qp[i] = (qr > 0.f) ? qr + 1.f : __expf(qr);   // elu(x)+1
    kp[i] = (kr > 0.f) ? kr + 1.f : __expf(kr);
}

// ---------------- sequential decayed linear-attention scan, one block per (b,h) ----------------
__global__ __launch_bounds__(64)
void scan_kernel(const float* __restrict__ qkv, const float* __restrict__ gate,
                 const float* __restrict__ head_decay, float* __restrict__ att) {
    const int bh = blockIdx.x;
    const int b  = bh / H_;
    const int h  = bh % H_;
    const int i  = threadIdx.x;          // owns KV column e=i and z row d=i

    const float dec   = fminf(fmaxf(0.3f + 0.65f * sigmoidf_(head_decay[h]), 1e-5f), 0.999f);
    const float one_m = 1.0f - dec;

    __shared__ __align__(16) float sq[DH_];
    __shared__ __align__(16) float sk[DH_];
    __shared__ __align__(16) float sz[DH_];
    float4 kv4[DH_ / 4];
#pragma unroll
    for (int d = 0; d < DH_ / 4; ++d) kv4[d] = make_float4(0.f, 0.f, 0.f, 0.f);
    float z = 0.f;

    const float4* sq4 = (const float4*)sq;
    const float4* sk4 = (const float4*)sk;
    const float4* sz4 = (const float4*)sz;

    for (int t = 0; t < L_; ++t) {
        const size_t row = (size_t)(b * L_ + t);
        const float* rp = qkv + row * (3 * D_) + h * DH_;
        sq[i] = rp[i];
        sk[i] = rp[D_ + i];
        const float ve = rp[2 * D_ + i];
        const float g  = gate[row * H_ + h];
        __syncthreads();

        z = dec * z + sk[i] * one_m;
        sz[i] = sq[i] * z;

        const float coef = ve * (g * one_m);
        float num = 0.f;
#pragma unroll
        for (int d = 0; d < DH_ / 4; ++d) {
            const float4 kk = sk4[d];
            const float4 qq = sq4[d];
            kv4[d].x = dec * kv4[d].x + kk.x * coef;  num += qq.x * kv4[d].x;
            kv4[d].y = dec * kv4[d].y + kk.y * coef;  num += qq.y * kv4[d].y;
            kv4[d].z = dec * kv4[d].z + kk.z * coef;  num += qq.z * kv4[d].z;
            kv4[d].w = dec * kv4[d].w + kk.w * coef;  num += qq.w * kv4[d].w;
        }
        __syncthreads();

        float den = 0.f;
#pragma unroll
        for (int d = 0; d < DH_ / 4; ++d) {
            const float4 zz = sz4[d];
            den += zz.x + zz.y + zz.z + zz.w;
        }
        den = fmaxf(den, 1e-5f);
        att[row * D_ + h * DH_ + i] = num / den;
    }
}

// ---------------- host ----------------
extern "C" void kernel_launch(void* const* d_in, const int* in_sizes, int n_in,
                              void* d_out, int out_size, void* d_ws, size_t ws_size,
                              hipStream_t stream) {
    (void)in_sizes; (void)n_in; (void)out_size; (void)ws_size;
    const float* x       = (const float*)d_in[0];
    const float* w_ln    = (const float*)d_in[1];
    const float* w_comp  = (const float*)d_in[2];
    const float* w_qkv   = (const float*)d_in[3];
    const float* w_reso  = (const float*)d_in[4];
    const float* w_qn    = (const float*)d_in[5];
    const float* w_kn    = (const float*)d_in[6];
    const float* h_decay = (const float*)d_in[7];
    const float* temper  = (const float*)d_in[8];
    const float* w_og    = (const float*)d_in[9];
    const float* w_proj  = (const float*)d_in[10];
    const float* w_mem   = (const float*)d_in[11];
    float* out = (float*)d_out;

    char* p = (char*)d_ws;
    auto alloc = [&](size_t bytes) -> void* {
        void* r = (void*)p;
        p += (bytes + 255) & ~(size_t)255;
        return r;
    };
    __bf16* xn_bf   = (__bf16*)alloc((size_t)MROWS * D_ * 2);
    __bf16* wc_bf   = (__bf16*)alloc((size_t)D_ * LAT_ * 2);
    __bf16* wqkv_bf = (__bf16*)alloc((size_t)LAT_ * 3 * D_ * 2);
    __bf16* wog_bf  = (__bf16*)alloc((size_t)LAT_ * D_ * 2);
    __bf16* wpr_bf  = (__bf16*)alloc((size_t)D_ * D_ * 2);
    __bf16* lat_bf  = (__bf16*)alloc((size_t)MROWS * LAT_ * 2);
    float*  qkv_f   = (float*)alloc((size_t)MROWS * 3 * D_ * 4);
    float*  gate_f  = (float*)alloc((size_t)MROWS * H_ * 4);
    float*  att_f   = (float*)alloc((size_t)MROWS * D_ * 4);
    __bf16* m_bf    = (__bf16*)alloc((size_t)MROWS * D_ * 2);
    float*  og_f    = (float*)alloc((size_t)MROWS * D_ * 4);

    auto cast = [&](const float* s, __bf16* d, int n) {
        cast_f32_bf16<<<(n + 255) / 256, 256, 0, stream>>>(s, d, n);
    };
    cast(w_comp, wc_bf, D_ * LAT_);
    cast(w_qkv,  wqkv_bf, LAT_ * 3 * D_);
    cast(w_og,   wog_bf, LAT_ * D_);
    cast(w_proj, wpr_bf, D_ * D_);

    rms_cast<<<MROWS, 256, 0, stream>>>(x, w_ln, xn_bf, D_);

    // latent = silu(xn @ w_compress)  -> bf16
    gemm_wmma_bf16<<<dim3(MROWS / 128, LAT_ / 64), 256, 0, stream>>>(
        xn_bf, wc_bf, nullptr, lat_bf, nullptr, MROWS, LAT_, D_, 1);
    // qkv = latent @ w_qkv -> f32
    gemm_wmma_bf16<<<dim3(MROWS / 128, (3 * D_) / 64), 256, 0, stream>>>(
        lat_bf, wqkv_bf, qkv_f, nullptr, nullptr, MROWS, 3 * D_, LAT_, 0);
    // resonance gate
    gate_kernel<<<(MROWS * H_ + 255) / 256, 256, 0, stream>>>(lat_bf, w_reso, temper, gate_f);
    // q/k head RMS + RoPE + elu+1 (in place)
    qk_transform<<<MROWS * H_, 64, 0, stream>>>(qkv_f, w_qn, w_kn);
    // decayed linear-attention scan
    scan_kernel<<<B_ * H_, 64, 0, stream>>>(qkv_f, gate_f, h_decay, att_f);
    // mem RMSNorm -> bf16
    rms_cast<<<MROWS, 256, 0, stream>>>(att_f, w_mem, m_bf, D_);
    // out_gate = silu(latent @ w_out_gate) -> f32
    gemm_wmma_bf16<<<dim3(MROWS / 128, D_ / 64), 256, 0, stream>>>(
        lat_bf, wog_bf, og_f, nullptr, nullptr, MROWS, D_, LAT_, 2);
    // final = (m @ w_proj) * out_gate -> d_out
    gemm_wmma_bf16<<<dim3(MROWS / 128, D_ / 64), 256, 0, stream>>>(
        m_bf, wpr_bf, out, nullptr, og_f, MROWS, D_, D_, 3);
}
